// GraphDiffusionModel_24876450578534
// MI455X (gfx1250) — compile-verified, matrix operand order checked by reading, not run on hardware
//
#include <hip/hip_runtime.h>
#include <hip/hip_bf16.h>

typedef __attribute__((ext_vector_type(16))) __bf16 v16bf;
typedef __attribute__((ext_vector_type(8)))  float  v8f;

static constexpr int NN    = 4096;
static constexpr int NE    = 131072;
static constexpr int NF    = 64;
static constexpr int EF    = 32;
static constexpr int HID   = 256;
static constexpr int HEADS = 8;
static constexpr int QKVW  = 768;
static constexpr int LAYERS= 4;

struct Seg { const __bf16* p; const int* idx; int stride; int kw; };

__device__ __forceinline__ float act_apply(float v, int act) {
  if (act == 1) return fmaxf(v, 0.f);
  if (act == 2) return v * (1.f / (1.f + __expf(-v)));   // silu
  return v;
}

// ---- CDNA5 async global->LDS copy (ASYNCcnt-tracked, no VGPR round trip) ----
__device__ __forceinline__ unsigned lds_off(const void* p) {
  return (unsigned)(unsigned long long)
      (__attribute__((address_space(3))) const char*)p;
}
// copy 32 bytes: two b128 transfers sharing one address pair (INST_OFFSET is
// applied to both the global and the LDS address per the ISA)
__device__ __forceinline__ void async_copy32(void* l, const void* g) {
  unsigned lo = lds_off(l);
  unsigned long long ga = (unsigned long long)g;
  asm volatile("global_load_async_to_lds_b128 %0, %1, off"
               :: "v"(lo), "v"(ga) : "memory");
  asm volatile("global_load_async_to_lds_b128 %0, %1, off offset:16"
               :: "v"(lo), "v"(ga) : "memory");
}
__device__ __forceinline__ void async_wait0() {
  asm volatile("s_wait_asynccnt 0x0" ::: "memory");
}

// ---------------------------------------------------------------------------
// Generic WMMA GEMM:  out[M,N] = act(A @ W^T + bias + colvec)
// A is a virtual [M,K] bf16 matrix made of up to 3 concatenated K-segments,
// each optionally row-gathered through an int32 index array.  Segment fields
// live in registers (uniform if/else, no address-of -> no scratch); gather
// indices are read from the LDS rowmap ONCE per segment and folded into a
// hoisted base pointer; tiles are staged with GLOBAL_LOAD_ASYNC_TO_LDS_B128.
// Tile: 64x64 per block, 4 waves, each wave owns a 16-row strip.
// ---------------------------------------------------------------------------
__global__ __launch_bounds__(128) void k_gemm(
    Seg s0, Seg s1, Seg s2, int nseg,
    const __bf16* __restrict__ W, const float* __restrict__ bias,
    const float* __restrict__ colvec, int act,
    float* __restrict__ outf, __bf16* __restrict__ outb,
    int M, int K, int N)
{
  __shared__ __bf16 As[64][40];          // 80B pitch = 5 x 16B (aligned b128)
  __shared__ __bf16 Ws[64][40];
  __shared__ int    rowmap[64];
  const int tid  = threadIdx.x;
  const int wave = tid >> 5, lane = tid & 31;
  const int hi   = lane >> 4, l15 = lane & 15;
  const int m0 = blockIdx.x * 64, n0 = blockIdx.y * 64;
  const int kb = hi * 8;
  const int lrow = tid >> 1, lh = tid & 1;   // staging unit: 32B of one row

  // W rows past N never change: zero them once, skip in the K loop.
  const bool wvalid = (n0 + lrow) < N;
  if (!wvalid) {
    uint4 z = make_uint4(0u, 0u, 0u, 0u);
    *(uint4*)&Ws[lrow][lh * 16]     = z;
    *(uint4*)&Ws[lrow][lh * 16 + 8] = z;
  }
  const __bf16* wbase = wvalid ? (W + (long)(n0 + lrow) * K + lh * 16) : nullptr;

  v8f acc[4] = {};

  int kbase = 0;
  for (int si = 0; si < nseg; ++si) {
    const __bf16* sp; const int* sidx; int sstride, skw;
    if (si == 0)      { sp = s0.p; sidx = s0.idx; sstride = s0.stride; skw = s0.kw; }
    else if (si == 1) { sp = s1.p; sidx = s1.idx; sstride = s1.stride; skw = s1.kw; }
    else              { sp = s2.p; sidx = s2.idx; sstride = s2.stride; skw = s2.kw; }

    if (tid < 64) rowmap[tid] = sidx ? sidx[m0 + tid] : (m0 + tid);
    __syncthreads();
    // hoist the gather: one LDS read per segment, folded into a base pointer
    const __bf16* abase = sp + (long)rowmap[lrow] * sstride + lh * 16;

    for (int kk0 = 0; kk0 < skw; kk0 += 32) {
      const int kglob = kbase + kk0;
      async_copy32(&As[lrow][lh * 16], abase + kk0);          // A tile 64x32
      if (wbase) async_copy32(&Ws[lrow][lh * 16], wbase + kglob); // W tile 64x32
      async_wait0();
      __syncthreads();

      v16bf a;
      #pragma unroll
      for (int i = 0; i < 8; ++i) {
        a[i]     = As[wave * 16 + l15][kb + i];
        a[i + 8] = As[wave * 16 + l15][kb + 16 + i];
      }
      v16bf bf[4];
      #pragma unroll
      for (int nt = 0; nt < 4; ++nt)
        #pragma unroll
        for (int i = 0; i < 8; ++i) {
          bf[nt][i]     = Ws[nt * 16 + l15][kb + i];
          bf[nt][i + 8] = Ws[nt * 16 + l15][kb + 16 + i];
        }
      #pragma unroll
      for (int nt = 0; nt < 4; ++nt)
        acc[nt] = __builtin_amdgcn_wmma_f32_16x16x32_bf16(
            false, a, false, bf[nt], (short)0, acc[nt], false, false);
      __syncthreads();
    }
    kbase += skw;
  }

  #pragma unroll
  for (int nt = 0; nt < 4; ++nt) {
    int c = n0 + nt * 16 + l15;
    if (c >= N) continue;
    float bv = bias ? bias[c] : 0.f;
    float cv = colvec ? colvec[c] : 0.f;
    #pragma unroll
    for (int r = 0; r < 8; ++r) {
      int row = m0 + wave * 16 + hi * 8 + r;
      float v = act_apply(acc[nt][r] + bv + cv, act);
      if (outf) outf[(long)row * N + c] = v;
      if (outb) outb[(long)row * N + c] = (__bf16)v;
    }
  }
}

// ---------------------------------------------------------------------------
// WMMA GEMM with fused row-LayerNorm epilogue.  BN == N (128 or 256), block
// tile 32xBN (whole output row in one block -> LN fuses).  Optional residual
// before LN; optional activation after LN; optional atomic scatter-add of the
// result to outf[scatter_idx[row]] (edge->node message aggregation).
// ---------------------------------------------------------------------------
template<int BN>
__global__ __launch_bounds__(256) void k_gemm_ln(
    Seg s0, Seg s1, Seg s2, int nseg,
    const __bf16* __restrict__ W, const float* __restrict__ bias,
    const float* __restrict__ residual,
    const float* __restrict__ ln_g, const float* __restrict__ ln_b,
    int act, const int* __restrict__ scatter_idx,
    float* __restrict__ outf, __bf16* __restrict__ outb, int K)
{
  __shared__ __bf16 As[32][40];
  __shared__ __bf16 Ws[BN][40];
  __shared__ float  Cs[32][BN + 1];
  __shared__ float  mu_s[32], rs_s[32];
  __shared__ int    rowmap[32];

  const int tid  = threadIdx.x;
  const int wave = tid >> 5, lane = tid & 31;
  const int hi   = lane >> 4, l15 = lane & 15;
  const int m0 = blockIdx.x * 32;
  const int mt = wave & 1;
  const int NCNT = BN / 64;                  // n-tiles per wave
  const int nb = (wave >> 1) * NCNT;
  const int kb = hi * 8;
  const int ar = tid >> 1, ah = tid & 1;     // A staging unit (tid < 64)

  // hoisted W base pointers (uniform layout, BN == N)
  const __bf16* wbase[BN / 128];
  #pragma unroll
  for (int u = 0; u < BN / 128; ++u) {
    int unit = tid + u * 256;
    wbase[u] = W + (long)(unit >> 1) * K + (unit & 1) * 16;
  }

  v8f acc[NCNT] = {};

  int kbase = 0;
  for (int si = 0; si < nseg; ++si) {
    const __bf16* sp; const int* sidx; int sstride, skw;
    if (si == 0)      { sp = s0.p; sidx = s0.idx; sstride = s0.stride; skw = s0.kw; }
    else if (si == 1) { sp = s1.p; sidx = s1.idx; sstride = s1.stride; skw = s1.kw; }
    else              { sp = s2.p; sidx = s2.idx; sstride = s2.stride; skw = s2.kw; }

    if (tid < 32) rowmap[tid] = sidx ? sidx[m0 + tid] : (m0 + tid);
    __syncthreads();
    const __bf16* abase =
        (tid < 64) ? (sp + (long)rowmap[ar] * sstride + ah * 16) : nullptr;

    for (int kk0 = 0; kk0 < skw; kk0 += 32) {
      const int kglob = kbase + kk0;
      if (abase) async_copy32(&As[ar][ah * 16], abase + kk0);   // A tile 32x32
      #pragma unroll
      for (int u = 0; u < BN / 128; ++u) {                      // W tile BNx32
        int unit = tid + u * 256;
        async_copy32(&Ws[unit >> 1][(unit & 1) * 16], wbase[u] + kglob);
      }
      async_wait0();
      __syncthreads();

      v16bf a;
      #pragma unroll
      for (int i = 0; i < 8; ++i) {
        a[i]     = As[mt * 16 + l15][kb + i];
        a[i + 8] = As[mt * 16 + l15][kb + 16 + i];
      }
      v16bf bf[NCNT];
      #pragma unroll
      for (int c = 0; c < NCNT; ++c)
        #pragma unroll
        for (int i = 0; i < 8; ++i) {
          bf[c][i]     = Ws[(nb + c) * 16 + l15][kb + i];
          bf[c][i + 8] = Ws[(nb + c) * 16 + l15][kb + 16 + i];
        }
      #pragma unroll
      for (int c = 0; c < NCNT; ++c)
        acc[c] = __builtin_amdgcn_wmma_f32_16x16x32_bf16(
            false, a, false, bf[c], (short)0, acc[c], false, false);
      __syncthreads();
    }
    kbase += skw;
  }

  // stage C (+bias, +residual) to LDS for the row-wise LayerNorm
  #pragma unroll
  for (int c = 0; c < NCNT; ++c) {
    int col = (nb + c) * 16 + l15;
    #pragma unroll
    for (int r = 0; r < 8; ++r) {
      int mrow = mt * 16 + hi * 8 + r;
      float v = acc[c][r] + bias[col];
      if (residual) v += residual[(long)(m0 + mrow) * BN + col];
      Cs[mrow][col] = v;
    }
  }
  __syncthreads();
  if (tid < 32) {
    float s = 0.f, s2 = 0.f;
    for (int j = 0; j < BN; ++j) { float x = Cs[tid][j]; s += x; s2 += x * x; }
    float mu = s / BN;
    float var = s2 / BN - mu * mu;
    mu_s[tid] = mu; rs_s[tid] = rsqrtf(var + 1e-5f);
  }
  __syncthreads();
  const int SH = (BN == 256) ? 8 : 7;
  #pragma unroll
  for (int i = 0; i < (32 * BN) / 256; ++i) {
    int e = tid + i * 256;
    int m = e >> SH, col = e & (BN - 1);
    float v = (Cs[m][col] - mu_s[m]) * rs_s[m] * ln_g[col] + ln_b[col];
    v = act_apply(v, act);
    long grow = m0 + m;
    if (scatter_idx) atomicAdd(&outf[(long)scatter_idx[grow] * BN + col], v);
    else {
      if (outf) outf[grow * BN + col] = v;
      if (outb) outb[grow * BN + col] = (__bf16)v;
    }
  }
}

// ---------------------------------------------------------------------------
// Flash attention.  dh=32 == one bf16 WMMA K-step.  Block = one head x 64
// query rows; streams 64-key blocks with online softmax; O lives in WMMA
// accumulators and is rescaled in-register each step.  Both GEMMs use WMMA.
// Q/K tiles staged with async DMA; V tile needs a transpose so it goes
// vector-load + scalar LDS store.
// ---------------------------------------------------------------------------
__global__ __launch_bounds__(128) void k_flash_attn(
    const __bf16* __restrict__ qkv, __bf16* __restrict__ outb, int N)
{
  __shared__ __bf16 Qs[64][40];
  __shared__ __bf16 Ks[64][40];
  __shared__ __bf16 VTs[32][68];
  __shared__ float  Sf[64][65];
  __shared__ __bf16 Pb[64][68];
  __shared__ float  mrow[64], lrow[64], arow[64];

  const int tid = threadIdx.x, wave = tid >> 5, lane = tid & 31;
  const int hi  = lane >> 4, l15 = lane & 15;
  const int h = blockIdx.y, m0 = blockIdx.x * 64, kb = hi * 8;
  const int ur = tid >> 1, uh = tid & 1;

  // hoisted per-thread base pointers into the packed qkv tensor
  const __bf16* kbase_p = qkv + (long)ur * QKVW + 256 + h * 32 + uh * 16;
  const int vk = tid >> 2, vd = (tid & 3) * 8;
  const __bf16* vbase_p = qkv + (long)vk * QKVW + 512 + h * 32 + vd;

  async_copy32(&Qs[ur][uh * 16],                       // Q tile via async DMA
               qkv + (long)(m0 + ur) * QKVW + h * 32 + uh * 16);
  if (tid < 64) { mrow[tid] = -3.0e38f; lrow[tid] = 0.f; }
  async_wait0();
  __syncthreads();

  v16bf aq;
  #pragma unroll
  for (int i = 0; i < 8; ++i) {
    aq[i]     = Qs[wave * 16 + l15][kb + i];
    aq[i + 8] = Qs[wave * 16 + l15][kb + 16 + i];
  }
  v8f o0 = {}, o1 = {};

  for (int kblk = 0; kblk < N; kblk += 64) {
    const long koff = (long)kblk * QKVW;
    async_copy32(&Ks[ur][uh * 16], kbase_p + koff);    // K tile via async DMA
    #pragma unroll
    for (int u = 0; u < 2; ++u) {                      // V tile: vector load, transposed store
      uint4 d = *(const uint4*)(vbase_p + koff + (long)u * 32 * QKVW);
      const __bf16* db = (const __bf16*)&d;
      #pragma unroll
      for (int j = 0; j < 8; ++j) VTs[vd + j][vk + u * 32] = db[j];
    }
    async_wait0();
    __syncthreads();

    v8f s[4] = {};
    {
      v16bf bf[4];
      #pragma unroll
      for (int nt = 0; nt < 4; ++nt)
        #pragma unroll
        for (int i = 0; i < 8; ++i) {
          bf[nt][i]     = Ks[nt * 16 + l15][kb + i];
          bf[nt][i + 8] = Ks[nt * 16 + l15][kb + 16 + i];
        }
      #pragma unroll
      for (int nt = 0; nt < 4; ++nt)
        s[nt] = __builtin_amdgcn_wmma_f32_16x16x32_bf16(
            false, aq, false, bf[nt], (short)0, s[nt], false, false);
    }
    #pragma unroll
    for (int nt = 0; nt < 4; ++nt)
      #pragma unroll
      for (int r = 0; r < 8; ++r)
        Sf[wave * 16 + hi * 8 + r][nt * 16 + l15] = s[nt][r] * 0.17677669529663687f;
    __syncthreads();

    if (tid < 64) {                                    // online softmax, row-owner
      int r = tid;
      float mold = mrow[r], mx = mold;
      for (int k = 0; k < 64; ++k) mx = fmaxf(mx, Sf[r][k]);
      float al = __expf(mold - mx), sum = 0.f;
      for (int k = 0; k < 64; ++k) {
        float p = __expf(Sf[r][k] - mx);
        Pb[r][k] = (__bf16)p; sum += p;
      }
      mrow[r] = mx; lrow[r] = al * lrow[r] + sum; arow[r] = al;
    }
    __syncthreads();

    #pragma unroll
    for (int r = 0; r < 8; ++r) {                      // rescale O accumulators
      float al = arow[wave * 16 + hi * 8 + r];
      o0[r] *= al; o1[r] *= al;
    }
    #pragma unroll
    for (int ks = 0; ks < 2; ++ks) {                   // O += P @ V
      v16bf pa;
      #pragma unroll
      for (int i = 0; i < 8; ++i) {
        pa[i]     = Pb[wave * 16 + l15][ks * 32 + kb + i];
        pa[i + 8] = Pb[wave * 16 + l15][ks * 32 + kb + 16 + i];
      }
      v16bf b0, b1;
      #pragma unroll
      for (int i = 0; i < 8; ++i) {
        b0[i]     = VTs[l15][ks * 32 + kb + i];
        b0[i + 8] = VTs[l15][ks * 32 + kb + 16 + i];
        b1[i]     = VTs[16 + l15][ks * 32 + kb + i];
        b1[i + 8] = VTs[16 + l15][ks * 32 + kb + 16 + i];
      }
      o0 = __builtin_amdgcn_wmma_f32_16x16x32_bf16(false, pa, false, b0, (short)0, o0, false, false);
      o1 = __builtin_amdgcn_wmma_f32_16x16x32_bf16(false, pa, false, b1, (short)0, o1, false, false);
    }
    __syncthreads();
  }

  #pragma unroll
  for (int r = 0; r < 8; ++r) {
    int row = wave * 16 + hi * 8 + r;
    float inv = 1.f / lrow[row];
    outb[(long)(m0 + row) * HID + h * 32 + l15]      = (__bf16)(o0[r] * inv);
    outb[(long)(m0 + row) * HID + h * 32 + 16 + l15] = (__bf16)(o1[r] * inv);
  }
}

// ------------------------------- small kernels -----------------------------
__global__ void k_f2b(const float* __restrict__ s, __bf16* __restrict__ d, long n) {
  for (long i = blockIdx.x * (long)blockDim.x + threadIdx.x; i < n;
       i += (long)gridDim.x * blockDim.x) d[i] = (__bf16)s[i];
}
__global__ void k_zero(float* p, long n) {
  for (long i = blockIdx.x * (long)blockDim.x + threadIdx.x; i < n;
       i += (long)gridDim.x * blockDim.x) p[i] = 0.f;
}
__global__ void k_time_embed(const float* t, const float* w1, const float* b1,
                             const float* w2, const float* b2, float* temb) {
  __shared__ float t1[HID];
  int i = threadIdx.x;
  float v = w1[i] * t[0] + b1[i];
  t1[i] = v * (1.f / (1.f + __expf(-v)));
  __syncthreads();
  float acc = b2[i];
  for (int j = 0; j < HID; ++j) acc += w2[i * HID + j] * t1[j];
  temb[i] = acc;
}
__global__ void k_pair_avg(const float* __restrict__ nh, const int* __restrict__ row,
                           const int* __restrict__ col, __bf16* __restrict__ out, long total) {
  for (long i = blockIdx.x * (long)blockDim.x + threadIdx.x; i < total;
       i += (long)gridDim.x * blockDim.x) {
    long e = i >> 8; int k = (int)(i & 255);
    out[i] = (__bf16)(0.5f * (nh[(long)row[e] * HID + k] + nh[(long)col[e] * HID + k]));
  }
}
__global__ void k_colmean(const float* x, float* g, int rows) {
  int j = threadIdx.x;
  float s = 0.f;
  for (int r = 0; r < rows; ++r) s += x[(long)r * HID + j];
  g[j] = s / rows;
}
__global__ void k_topo1(const float* g, const float* w, const float* b, float* t1) {
  __shared__ float gs[HID];
  for (int i = threadIdx.x; i < HID; i += blockDim.x) gs[i] = g[i];
  __syncthreads();
  int o = threadIdx.x;
  float acc = b[o];
  for (int j = 0; j < HID; ++j) acc += w[o * HID + j] * gs[j];
  t1[o] = fmaxf(acc, 0.f);
}
__global__ void k_topo2(const float* t1, const float* w, const float* b, float* out, int n) {
  __shared__ float ts[128];
  for (int i = threadIdx.x; i < 128; i += blockDim.x) ts[i] = t1[i];
  __syncthreads();
  int o = blockIdx.x * blockDim.x + threadIdx.x;
  if (o >= n) return;
  float acc = b[o];
  for (int j = 0; j < 128; ++j) acc += w[o * 128 + j] * ts[j];
  out[o] = acc;
}

// --------------------------- input index map -------------------------------
// jax pytree (sorted-key) flatten of setup_inputs():
enum {
  IN_NODE_F = 0, IN_EDGE_F = 1, IN_GLOB_F = 2, IN_T = 3,
  P_EO_L1_B = 4, P_EO_L1_W, P_EO_L2_B, P_EO_L2_W, P_EO_LN_B, P_EO_LN_G,
  P_EPROJ_B, P_EPROJ_W, P_GPROJ_B, P_GPROJ_W,
  P_LAYER0 = 14,   // +12/layer: in_b,in_w,out_b,out_w, eln_b,eln_g, emlp_b,emlp_w, ln_b,ln_g, oproj_b,oproj_w
  P_NO_L1_B = 62, P_NO_L1_W, P_NO_L2_B, P_NO_L2_W, P_NO_LN_B, P_NO_LN_G,
  P_NPROJ_B, P_NPROJ_W, P_T1_B, P_T1_W, P_T2_B, P_T2_W,
  P_TOPO1_B, P_TOPO1_W, P_TOPO2_B, P_TOPO2_W,
  IN_EDGE_INDEX = 78
};

static inline Seg mkseg(const __bf16* p, const int* idx, int stride, int kw) {
  Seg s; s.p = p; s.idx = idx; s.stride = stride; s.kw = kw; return s;
}

extern "C" void kernel_launch(void* const* d_in, const int* in_sizes, int n_in,
                              void* d_out, int out_size, void* d_ws, size_t ws_size,
                              hipStream_t stream)
{
  (void)in_sizes; (void)n_in; (void)out_size; (void)ws_size;
  auto F = [&](int i) { return (const float*)d_in[i]; };
  const int* eidx = (const int*)d_in[IN_EDGE_INDEX];
  const int* row = eidx;
  const int* col = eidx + NE;

  // ---- workspace carve ----
  char* ws = (char*)d_ws;
  size_t off = 0;
  auto alloc = [&](size_t bytes) -> char* {
    char* p = ws + off; off += (bytes + 255) & ~(size_t)255; return p;
  };
  float* t_emb    = (float*)alloc(HID * 4);
  float* node_h   = (float*)alloc((size_t)NN * HID * 4);
  float* node_msg = (float*)alloc((size_t)NN * HID * 4);
  float* g_mean   = (float*)alloc(HID * 4);
  float* topo_t1  = (float*)alloc(128 * 4);
  __bf16* xb       = (__bf16*)alloc((size_t)NN * HID * 2);
  __bf16* xattn_bf = (__bf16*)alloc((size_t)NN * HID * 2);
  __bf16* attn_bf  = (__bf16*)alloc((size_t)NN * HID * 2);
  __bf16* nmsg_bf  = (__bf16*)alloc((size_t)NN * HID * 2);
  __bf16* qkv_bf   = (__bf16*)alloc((size_t)NN * QKVW * 2);
  __bf16* nodef_bf = (__bf16*)alloc((size_t)NN * NF * 2);
  __bf16* edgef_bf = (__bf16*)alloc((size_t)NE * EF * 2);
  __bf16* eb       = (__bf16*)alloc((size_t)NE * HID * 2);
  __bf16* pair_bf  = (__bf16*)alloc((size_t)NE * HID * 2);
  __bf16* head_bf  = (__bf16*)alloc((size_t)NE * 128 * 2);
  __bf16* w_nproj = (__bf16*)alloc(256 * 64 * 2);
  __bf16* w_eproj = (__bf16*)alloc(256 * 32 * 2);
  __bf16 *w_in[LAYERS], *w_out[LAYERS], *w_emlp[LAYERS], *w_oproj[LAYERS];
  for (int L = 0; L < LAYERS; ++L) {
    w_in[L]    = (__bf16*)alloc(768 * 256 * 2);
    w_out[L]   = (__bf16*)alloc(256 * 256 * 2);
    w_emlp[L]  = (__bf16*)alloc(256 * 768 * 2);
    w_oproj[L] = (__bf16*)alloc(256 * 512 * 2);
  }
  __bf16* w_no1 = (__bf16*)alloc(128 * 256 * 2);
  __bf16* w_no2 = (__bf16*)alloc(64 * 128 * 2);
  __bf16* w_eo1 = (__bf16*)alloc(128 * 256 * 2);
  __bf16* w_eo2 = (__bf16*)alloc(32 * 128 * 2);

  float* o = (float*)d_out;
  float* out_node_noise = o;                               // [4096,64]
  float* out_edge_noise = o + (size_t)NN * NF;             // [131072,32]
  float* out_topo       = out_edge_noise + (size_t)NE * EF;// [2500]
  float* out_nodeh      = out_topo + 2500;                 // [4096,256]

  auto conv = [&](const float* s, __bf16* d, long n) {
    long b = (n + 255) / 256; if (b > 8192) b = 8192;
    k_f2b<<<dim3((unsigned)b), dim3(256), 0, stream>>>(s, d, n);
  };
  Seg S0 = mkseg(nullptr, nullptr, 0, 0);

  // ---- bf16 conversions (inputs + weights), deterministic each call ----
  conv(F(IN_NODE_F), nodef_bf, (long)NN * NF);
  conv(F(IN_EDGE_F), edgef_bf, (long)NE * EF);
  conv(F(P_NPROJ_W), w_nproj, 256 * 64);
  conv(F(P_EPROJ_W), w_eproj, 256 * 32);
  for (int L = 0; L < LAYERS; ++L) {
    int base = P_LAYER0 + 12 * L;
    conv(F(base + 1),  w_in[L],    768 * 256);
    conv(F(base + 3),  w_out[L],   256 * 256);
    conv(F(base + 7),  w_emlp[L],  256 * 768);
    conv(F(base + 11), w_oproj[L], 256 * 512);
  }
  conv(F(P_NO_L1_W), w_no1, 128 * 256);
  conv(F(P_NO_L2_W), w_no2, 64 * 128);
  conv(F(P_EO_L1_W), w_eo1, 128 * 256);
  conv(F(P_EO_L2_W), w_eo2, 32 * 128);

  // ---- time embedding ----
  k_time_embed<<<1, HID, 0, stream>>>(F(IN_T), F(P_T1_W), F(P_T1_B),
                                      F(P_T2_W), F(P_T2_B), t_emb);

  // ---- node / edge projections ----
  k_gemm<<<dim3(NN / 64, HID / 64), 128, 0, stream>>>(
      mkseg(nodef_bf, nullptr, NF, NF), S0, S0, 1,
      w_nproj, F(P_NPROJ_B), t_emb, 0, node_h, xb, NN, NF, HID);
  k_gemm<<<dim3(NE / 64, HID / 64), 128, 0, stream>>>(
      mkseg(edgef_bf, nullptr, EF, EF), S0, S0, 1,
      w_eproj, F(P_EPROJ_B), nullptr, 0, nullptr, eb, NE, EF, HID);

  // ---- graph transformer layers ----
  for (int L = 0; L < LAYERS; ++L) {
    int base = P_LAYER0 + 12 * L;
    // qkv = x @ in_w^T + in_b
    k_gemm<<<dim3(NN / 64, QKVW / 64), 128, 0, stream>>>(
        mkseg(xb, nullptr, HID, HID), S0, S0, 1,
        w_in[L], F(base + 0), nullptr, 0, nullptr, qkv_bf, NN, HID, QKVW);
    // flash self-attention -> attn_bf
    k_flash_attn<<<dim3(NN / 64, HEADS), 128, 0, stream>>>(qkv_bf, attn_bf, NN);
    // x_attn = attn @ out_w^T + out_b
    k_gemm<<<dim3(NN / 64, HID / 64), 128, 0, stream>>>(
        mkseg(attn_bf, nullptr, HID, HID), S0, S0, 1,
        w_out[L], F(base + 2), nullptr, 0, nullptr, xattn_bf, NN, HID, HID);
    // node_msg = 0
    k_zero<<<dim3(2048), 256, 0, stream>>>(node_msg, (long)NN * HID);
    // msg = relu(LN(edge_mlp([x[row]‖x[col]‖edge_h]))) scatter-added to node_msg
    k_gemm_ln<256><<<dim3(NE / 32), 256, 0, stream>>>(
        mkseg(xb, row, HID, HID), mkseg(xb, col, HID, HID),
        mkseg(eb, nullptr, HID, HID), 3,
        w_emlp[L], F(base + 6), nullptr,
        F(base + 5), F(base + 4), /*relu*/1, col, node_msg, nullptr, 3 * HID);
    conv(node_msg, nmsg_bf, (long)NN * HID);
    // node_h = LN(out_proj([x_attn‖node_msg]) + node_h)
    k_gemm_ln<256><<<dim3(NN / 32), 256, 0, stream>>>(
        mkseg(xattn_bf, nullptr, HID, HID), mkseg(nmsg_bf, nullptr, HID, HID), S0, 2,
        w_oproj[L], F(base + 10), node_h,
        F(base + 9), F(base + 8), 0, nullptr, node_h, xb, 2 * HID);
  }

  // ---- node head: l2(silu(LN(l1(x)))) ----
  k_gemm_ln<128><<<dim3(NN / 32), 256, 0, stream>>>(
      mkseg(xb, nullptr, HID, HID), S0, S0, 1,
      w_no1, F(P_NO_L1_B), nullptr, F(P_NO_LN_G), F(P_NO_LN_B),
      /*silu*/2, nullptr, nullptr, head_bf, HID);
  k_gemm<<<dim3(NN / 64, 1), 128, 0, stream>>>(
      mkseg(head_bf, nullptr, 128, 128), S0, S0, 1,
      w_no2, F(P_NO_L2_B), nullptr, 0, out_node_noise, nullptr, NN, 128, NF);

  // ---- edge head on 0.5*(x[row]+x[col]) ----
  k_pair_avg<<<dim3(8192), 256, 0, stream>>>(node_h, row, col, pair_bf, (long)NE * HID);
  k_gemm_ln<128><<<dim3(NE / 32), 256, 0, stream>>>(
      mkseg(pair_bf, nullptr, HID, HID), S0, S0, 1,
      w_eo1, F(P_EO_L1_B), nullptr, F(P_EO_LN_G), F(P_EO_LN_B),
      /*silu*/2, nullptr, nullptr, head_bf, HID);
  k_gemm<<<dim3(NE / 64, 1), 128, 0, stream>>>(
      mkseg(head_bf, nullptr, 128, 128), S0, S0, 1,
      w_eo2, F(P_EO_L2_B), nullptr, 0, out_edge_noise, nullptr, NE, 128, EF);

  // ---- topo head ----
  k_colmean<<<1, HID, 0, stream>>>(node_h, g_mean, NN);
  k_topo1<<<1, 128, 0, stream>>>(g_mean, F(P_TOPO1_W), F(P_TOPO1_B), topo_t1);
  k_topo2<<<dim3(20), 128, 0, stream>>>(topo_t1, F(P_TOPO2_W), F(P_TOPO2_B), out_topo, 2500);

  // ---- node_h output ----
  hipMemcpyAsync(out_nodeh, node_h, (size_t)NN * HID * 4,
                 hipMemcpyDeviceToDevice, stream);
}
